// RGAT_16269336117432
// MI455X (gfx1250) — compile-verified, matrix operand order checked by reading, not run on hardware
//
#include <hip/hip_runtime.h>

typedef __attribute__((ext_vector_type(16))) __bf16 v16bf;
typedef __attribute__((ext_vector_type(8)))  float  v8f;

#define NEG_SLOPE 0.2f
#define RR 8
#define BB 4
#define COUT 64

// ---------- helpers ----------
__device__ inline unsigned short f2bf(float f) {
  unsigned u = __float_as_uint(f);
  unsigned r = u + 0x7FFFu + ((u >> 16) & 1u);  // round-to-nearest-even
  return (unsigned short)(r >> 16);
}
__device__ inline float bflo(unsigned u) { return __uint_as_float(u << 16); }
__device__ inline float bfhi(unsigned u) { return __uint_as_float(u & 0xFFFF0000u); }
// orderable encoding of float for unsigned atomicMax
__device__ inline unsigned encf(float f) {
  unsigned u = __float_as_uint(f);
  return (u & 0x80000000u) ? ~u : (u | 0x80000000u);
}
__device__ inline float decf(unsigned k) {
  unsigned u = (k & 0x80000000u) ? (k & 0x7FFFFFFFu) : ~k;
  return __uint_as_float(u);
}

// ---------- small kernels ----------
__global__ void k_f32_to_bf16(const float* __restrict__ in,
                              unsigned short* __restrict__ out, int n) {
  int i = blockIdx.x * blockDim.x + threadIdx.x;
  if (i < n) out[i] = f2bf(in[i]);
}

// w[r,i,o] = sum_b att[r,b] * basis[b,i,o]; store transposed bf16: wbT[r][o][i]
__global__ void k_make_w(const float* __restrict__ basis, const float* __restrict__ att,
                         unsigned short* __restrict__ wbT, int cin) {
  int idx = blockIdx.x * blockDim.x + threadIdx.x;
  int tot = RR * cin * COUT;
  if (idx >= tot) return;
  int o = idx % COUT;
  int i = (idx / COUT) % cin;
  int r = idx / (COUT * cin);
  float acc = 0.f;
#pragma unroll
  for (int b = 0; b < BB; ++b)
    acc += att[r * BB + b] * basis[((size_t)b * cin + i) * COUT + o];
  wbT[((size_t)r * COUT + o) * cin + i] = f2bf(acc);
}

// ---------- WMMA GEMM: xall[r][n][o] = sum_i xb[n][i] * w[r][i][o] (bf16 in/out, f32 acc) ----------
union AFrag { v16bf v; unsigned u[8]; };

__global__ __launch_bounds__(256)
void k_gemm_xall(const unsigned short* __restrict__ xb,
                 const unsigned short* __restrict__ wbT,
                 unsigned short* __restrict__ xall, int N, int cin) {
  const int r    = blockIdx.y;
  const int wave = threadIdx.x >> 5;
  const int lane = threadIdx.x & 31;
  const int row0 = blockIdx.x * 128 + wave * 16;
  if (row0 >= N) return;                 // wave-uniform: EXEC all-ones inside
  const int m    = lane & 15;
  const int half = lane >> 4;
  v8f acc[4] = {};
  const unsigned short* xrow  = xb + (size_t)(row0 + m) * cin;
  const unsigned short* wbase = wbT + (size_t)r * COUT * cin;
  for (int kb = 0; kb < cin; kb += 32) {
    // A fragment: 16x32 bf16 (lane = M, VGPR/half = K pairs per ISA layout)
    AFrag a;
#pragma unroll
    for (int v = 0; v < 8; ++v) {
      int k = ((v >> 2) << 4) + (half << 3) + ((v & 3) << 1);
      a.u[v] = *(const unsigned*)(xrow + kb + k);
    }
#pragma unroll
    for (int t = 0; t < 4; ++t) {
      // B fragment: 32x16 bf16, lane = N (output col), same K pattern; wbT rows contiguous in k
      AFrag bf;
      const unsigned short* wrow = wbase + (size_t)(t * 16 + m) * cin + kb;
#pragma unroll
      for (int v = 0; v < 8; ++v) {
        int k = ((v >> 2) << 4) + (half << 3) + ((v & 3) << 1);
        bf.u[v] = *(const unsigned*)(wrow + k);
      }
      acc[t] = __builtin_amdgcn_wmma_f32_16x16x32_bf16(
          false, a.v, false, bf.v, (short)0, acc[t], false, false);
    }
  }
  // C/D layout: col = lane&15, row = VGPR idx + 8*(lane>=16)
#pragma unroll
  for (int t = 0; t < 4; ++t) {
#pragma unroll
    for (int v = 0; v < 8; ++v) {
      int row = row0 + v + half * 8;
      int col = t * 16 + m;
      xall[((size_t)r * N + row) * COUT + col] = f2bf(acc[t][v]);
    }
  }
}

// xq[r,n] = dot(xall[r,n,:], q); xk likewise
__global__ void k_xqk(const unsigned short* __restrict__ xall,
                      const float* __restrict__ qv, const float* __restrict__ kv,
                      float* __restrict__ xq, float* __restrict__ xk, int N) {
  int idx = blockIdx.x * blockDim.x + threadIdx.x;
  if (idx >= RR * N) return;
  const unsigned* xa = (const unsigned*)(xall + (size_t)idx * COUT);
  float sq = 0.f, sk = 0.f;
#pragma unroll
  for (int j = 0; j < COUT / 2; ++j) {
    unsigned u = xa[j];
    float v0 = bflo(u), v1 = bfhi(u);
    sq += v0 * qv[2 * j] + v1 * qv[2 * j + 1];
    sk += v0 * kv[2 * j] + v1 * kv[2 * j + 1];
  }
  xq[idx] = sq;
  xk[idx] = sk;
}

__global__ void k_init(float* __restrict__ out, unsigned* __restrict__ mmax,
                       float* __restrict__ ssum, int N) {
  int i = blockIdx.x * blockDim.x + threadIdx.x;
  if (i < N * COUT) out[i] = 0.f;
  if (i < N) { mmax[i] = 0u; ssum[i] = 0.f; }
}

__global__ void k_edge_logit_max(const int* __restrict__ src, const int* __restrict__ dst,
                                 const int* __restrict__ et,
                                 const float* __restrict__ xq, const float* __restrict__ xk,
                                 float* __restrict__ ee, unsigned* __restrict__ mmax,
                                 int N, int E) {
  int e = blockIdx.x * blockDim.x + threadIdx.x;
  if (e >= E) return;
  int r = et[e], s = src[e], d = dst[e];
  float a = xq[(size_t)r * N + d] + xk[(size_t)r * N + s];
  a = (a > 0.f) ? a : NEG_SLOPE * a;  // leaky relu
  ee[e] = a;
  atomicMax(&mmax[d], encf(a));
}

__global__ void k_edge_exp(const int* __restrict__ dst, float* __restrict__ ee,
                           const unsigned* __restrict__ mmax, float* __restrict__ ssum,
                           int E) {
  int e = blockIdx.x * blockDim.x + threadIdx.x;
  if (e >= E) return;
  int d = dst[e];
  float m = decf(mmax[d]);
  float ex = __expf(ee[e] - m);
  ee[e] = ex;
  atomicAdd(&ssum[d], ex);
}

// one wave per edge; lane handles a packed bf16 pair of the 64-wide message
__global__ __launch_bounds__(256)
void k_edge_msg(const int* __restrict__ src, const int* __restrict__ dst,
                const int* __restrict__ et, const float* __restrict__ ee,
                const float* __restrict__ ssum, const unsigned short* __restrict__ xall,
                float* __restrict__ out, int N, int E) {
  int e = blockIdx.x * 8 + (int)(threadIdx.x >> 5);
  if (e >= E) return;
  int lane = threadIdx.x & 31;
  int d = dst[e], s = src[e], r = et[e];
  float coeff = ee[e] / (ssum[d] + 1e-16f);
  unsigned u = *(const unsigned*)(xall + ((size_t)r * N + s) * COUT + lane * 2);
  float* op = out + (size_t)d * COUT + lane * 2;
  atomicAdd(op,     coeff * bflo(u));
  atomicAdd(op + 1, coeff * bfhi(u));
}

__global__ void k_bias_act(float* __restrict__ out, const float* __restrict__ bias,
                           int N, int relu) {
  int i = blockIdx.x * blockDim.x + threadIdx.x;
  if (i >= N * COUT) return;
  float v = out[i] + bias[i & (COUT - 1)];
  if (relu) v = fmaxf(v, 0.f);
  out[i] = v;
}

// ---------- launcher ----------
extern "C" void kernel_launch(void* const* d_in, const int* in_sizes, int n_in,
                              void* d_out, int out_size, void* d_ws, size_t ws_size,
                              hipStream_t stream) {
  (void)n_in; (void)out_size; (void)ws_size;
  const float* x     = (const float*)d_in[0];
  const int*   eidx  = (const int*)d_in[1];
  const int*   etype = (const int*)d_in[2];
  const int N = in_sizes[0] / 128;
  const int E = in_sizes[2];
  const int* src = eidx;
  const int* dst = eidx + E;

  // workspace carve (256B aligned)
  char* p = (char*)d_ws;
  auto carve = [&](size_t bytes) -> void* {
    void* q = (void*)p;
    p += (bytes + 255) & ~(size_t)255;
    return q;
  };
  unsigned short* xb   = (unsigned short*)carve((size_t)N * 128 * 2);
  unsigned short* wbT  = (unsigned short*)carve((size_t)RR * COUT * 128 * 2);
  unsigned short* xall = (unsigned short*)carve((size_t)RR * N * COUT * 2);
  float*    xq   = (float*)carve((size_t)RR * N * 4);
  float*    xk   = (float*)carve((size_t)RR * N * 4);
  unsigned* mmax = (unsigned*)carve((size_t)N * 4);
  float*    ssum = (float*)carve((size_t)N * 4);
  float*    ee   = (float*)carve((size_t)E * 4);
  float*    hbuf = (float*)carve((size_t)N * COUT * 4);

  const int T = 256;
  const float* cur = x;
  int cin = 128;
  for (int l = 0; l < 3; ++l) {
    const float* basis = (const float*)d_in[3 + 5 * l];
    const float* att   = (const float*)d_in[4 + 5 * l];
    const float* qv    = (const float*)d_in[5 + 5 * l];
    const float* kv    = (const float*)d_in[6 + 5 * l];
    const float* bias  = (const float*)d_in[7 + 5 * l];
    float* out = (l == 2) ? (float*)d_out : hbuf;

    int nx = N * cin;
    k_f32_to_bf16<<<(nx + T - 1) / T, T, 0, stream>>>(cur, xb, nx);
    int nw = RR * cin * COUT;
    k_make_w<<<(nw + T - 1) / T, T, 0, stream>>>(basis, att, wbT, cin);
    dim3 gg((N + 127) / 128, RR);
    k_gemm_xall<<<gg, T, 0, stream>>>(xb, wbT, xall, N, cin);
    k_xqk<<<(RR * N + T - 1) / T, T, 0, stream>>>(xall, qv, kv, xq, xk, N);
    k_init<<<(N * COUT + T - 1) / T, T, 0, stream>>>(out, mmax, ssum, N);
    k_edge_logit_max<<<(E + T - 1) / T, T, 0, stream>>>(src, dst, etype, xq, xk, ee, mmax, N, E);
    k_edge_exp<<<(E + T - 1) / T, T, 0, stream>>>(dst, ee, mmax, ssum, E);
    k_edge_msg<<<(E + 7) / 8, T, 0, stream>>>(src, dst, etype, ee, ssum, xall, out, N, E);
    k_bias_act<<<(N * COUT + T - 1) / T, T, 0, stream>>>(out, bias, N, (l < 2) ? 1 : 0);

    cur = out;
    cin = COUT;
  }
}
